// Mamba_60455959658701
// MI455X (gfx1250) — compile-verified, matrix operand order checked by reading, not run on hardware
//
#include <hip/hip_runtime.h>
#include <hip/hip_bf16.h>

// ---------------------------------------------------------------------------
// Bidirectional Mamba block for MI455X (gfx1250, wave32, WMMA).
//
//   0. convert weight matrices f32 -> bf16 (padded to 64-row tiles, into ws)
//   1. gemm (bf16 WMMA): xz[b,e,l]    = in_proj_w . hs
//   2. conv+silu         xc[b,d,l]    (dir-aware causal depthwise conv)
//   3. gemm (bf16 WMMA): xdbl[b,96,l] = x_proj_w . xc
//   4. gemm (bf16 WMMA): dt[b,d,l]    = softplus(dt_proj_w . dtv + bias)
//   5. scan:             y[b,d,l]     (16-state recurrence, silu(z) gate)
//   (repeat 2-5 for backward dir; backward scan accumulates reversed)
//   6. gemm (bf16 WMMA): out[b,l,o]   = out_proj_w . y
//
// A-tiles are staged into LDS with CDNA5 async global->LDS b128 copies
// (weights pre-converted to bf16 and zero-padded so there is NO guarded
// staging path); B-tiles are staged with vectorized f32 loads + packed bf16
// conversion. Inner loop: 6x ds_load_b128 -> 2x v_wmma_f32_16x16x32_bf16.
// ---------------------------------------------------------------------------

#define BDIM     2
#define LSEQ     2048
#define DMODEL   1024
#define DSTATE   16
#define DCONV    4
#define DINNER   2048
#define DTRANK   64
#define XDBLROWS 96   // DTRANK + 2*DSTATE
#define XDBLPAD  128  // XDBLROWS padded to 64-row tile

typedef __attribute__((ext_vector_type(16))) __bf16 v16bf;
typedef __attribute__((ext_vector_type(4)))  __bf16 v4bf;
typedef __attribute__((ext_vector_type(8)))  float  v8f;
typedef __attribute__((ext_vector_type(4)))  int    v4i;

#if defined(__gfx1250__) && __has_builtin(__builtin_amdgcn_global_load_async_to_lds_b128)
#define HAVE_ASYNC_LDS 1
typedef __attribute__((address_space(1))) v4i GV4;  // global int4*
typedef __attribute__((address_space(3))) v4i LV4;  // LDS int4*
#else
#define HAVE_ASYNC_LDS 0
#endif

// ---------------- f32 -> bf16 convert (zero-pads [n4src, n4dst)) -----------
__global__ __launch_bounds__(256) void mamba_cvt_bf16(
    const float* __restrict__ src, __bf16* __restrict__ dst,
    long n4src, long n4dst) {
  const long i = (long)blockIdx.x * 256 + threadIdx.x;
  if (i < n4dst) {
    v4bf o = {(__bf16)0.f, (__bf16)0.f, (__bf16)0.f, (__bf16)0.f};
    if (i < n4src) {
      const float4 v = ((const float4*)src)[i];
      o[0] = (__bf16)v.x; o[1] = (__bf16)v.y;
      o[2] = (__bf16)v.z; o[3] = (__bf16)v.w;
    }
    ((v4bf*)dst)[i] = o;
  }
}

// ------------------------------ GEMM (WMMA) --------------------------------
// C[m,n] = sum_k A[m,k] * B(n,k)
//   A:  bf16, row-major MpadxK (K contiguous), Mpad = gridDim.x*BM (padded)
//   B:  f32,  Bp[(n/NB)*bBatch + (n%NB)*bStrideN + k*bStrideK]
//   C:  f32,  C[m*outStrideM + (n/NB)*outBatch + (n%NB)*outStrideN], m < M
// epilogue: 0 = none, 1 = softplus(x + bias[m])
// Requirements (all launches satisfy): K % 32 == 0, N % 64 == 0, NB % 64 == 0.
#define BM 64
#define BN 64
#define BK 32

__device__ __forceinline__ float softplus_f(float x) {
  return (x > 20.0f) ? x : log1pf(__expf(x));
}

__global__ __launch_bounds__(256) void mamba_gemm_wmma(
    const __bf16* __restrict__ A,
    const float* __restrict__ Bp, long bBatch, long bStrideN, long bStrideK,
    float* __restrict__ C, long outBatch, long outStrideM, long outStrideN,
    int M, int N, int K, int NB,
    int epilogue, const float* __restrict__ bias) {
  __shared__ __bf16 As[BM][BK];
  __shared__ __bf16 Bs[BN][BK];

  const int tid  = threadIdx.x;
  const int wave = tid >> 5;
  const int lane = tid & 31;
  const int g    = lane >> 4;
  const int ln16 = lane & 15;

  const int m0 = blockIdx.x * BM;
  const int n0 = blockIdx.y * BN;

  const int wm  = wave & 3;        // wave's m-tile (0..3)
  const int wn2 = (wave >> 2) * 2; // wave's pair of n-tiles

  const bool bKcontig = (bStrideK == 1);  // block-uniform

  v8f acc0 = {0.f, 0.f, 0.f, 0.f, 0.f, 0.f, 0.f, 0.f};
  v8f acc1 = {0.f, 0.f, 0.f, 0.f, 0.f, 0.f, 0.f, 0.f};

  // staging coordinates (row-major style: 4 threads x 8 elems per row)
  const int srow = tid >> 2;        // 0..63
  const int scol = (tid & 3) * 8;   // 0,8,16,24
  // staging coordinates (k-major style for strided-K B operands)
  const int sk = tid & 31;          // 0..31
  const int sn = (tid >> 5) * 8;    // 0,8,..,56

  for (int k0 = 0; k0 < K; k0 += BK) {
    // -------- stage A tile: unconditional async b128 copy (bf16 source) ----
    {
      const __bf16* gp = A + (long)(m0 + srow) * K + k0 + scol;
#if HAVE_ASYNC_LDS
      __builtin_amdgcn_global_load_async_to_lds_b128(
          (GV4*)gp, (LV4*)&As[srow][scol], 0, 0);
#else
      *(uint4*)&As[srow][scol] = *(const uint4*)gp;
#endif
    }

    // ---------------- stage B tile (f32 -> bf16, vectorized) --------------
    if (bKcontig) {
      const int  gn = n0 + srow;
      const long nb = gn / NB, nr = gn % NB;
      const float* gp = Bp + nb * bBatch + nr * bStrideN + k0 + scol;
      const float4 v0 = *(const float4*)gp;
      const float4 v1 = *(const float4*)(gp + 4);
      v4bf o0 = {(__bf16)v0.x, (__bf16)v0.y, (__bf16)v0.z, (__bf16)v0.w};
      v4bf o1 = {(__bf16)v1.x, (__bf16)v1.y, (__bf16)v1.z, (__bf16)v1.w};
      *(v4bf*)&Bs[srow][scol]     = o0;
      *(v4bf*)&Bs[srow][scol + 4] = o1;
    } else {  // bStrideN == 1: contiguous along n for fixed k
      const int  gn0 = n0 + sn;
      const long nb = gn0 / NB, nr = gn0 % NB;
      const float* gp = Bp + nb * bBatch + nr + (long)(k0 + sk) * bStrideK;
      const float4 v0 = *(const float4*)gp;
      const float4 v1 = *(const float4*)(gp + 4);
      Bs[sn + 0][sk] = (__bf16)v0.x;
      Bs[sn + 1][sk] = (__bf16)v0.y;
      Bs[sn + 2][sk] = (__bf16)v0.z;
      Bs[sn + 3][sk] = (__bf16)v0.w;
      Bs[sn + 4][sk] = (__bf16)v1.x;
      Bs[sn + 5][sk] = (__bf16)v1.y;
      Bs[sn + 6][sk] = (__bf16)v1.z;
      Bs[sn + 7][sk] = (__bf16)v1.w;
    }

#if HAVE_ASYNC_LDS
#if __has_builtin(__builtin_amdgcn_s_wait_asynccnt)
    __builtin_amdgcn_s_wait_asynccnt(0);
#else
    asm volatile("s_wait_asynccnt 0" ::: "memory");
#endif
#endif
    __syncthreads();

    // ---- A fragment per documented 16-bit 16x32 layout ----
    const int mr = wm * 16 + ln16;
    v16bf af;
#pragma unroll
    for (int v = 0; v < 8; ++v) {
      const int kb = (v < 4) ? (g * 8 + 2 * v) : (16 + g * 8 + 2 * (v - 4));
      af[2 * v]     = As[mr][kb];
      af[2 * v + 1] = As[mr][kb + 1];
    }
    // ---- two B fragments -> two WMMAs ----
#pragma unroll
    for (int t = 0; t < 2; ++t) {
      const int nr = (wn2 + t) * 16 + ln16;
      v16bf bfv;
#pragma unroll
      for (int v = 0; v < 8; ++v) {
        const int kb = (v < 4) ? (g * 8 + 2 * v) : (16 + g * 8 + 2 * (v - 4));
        bfv[2 * v]     = Bs[nr][kb];
        bfv[2 * v + 1] = Bs[nr][kb + 1];
      }
      if (t == 0) {
        acc0 = __builtin_amdgcn_wmma_f32_16x16x32_bf16(
            false, af, false, bfv, (short)0, acc0, false, false);
      } else {
        acc1 = __builtin_amdgcn_wmma_f32_16x16x32_bf16(
            false, af, false, bfv, (short)0, acc1, false, false);
      }
    }
    __syncthreads();
  }

  // ---- store: C/D 16x16 f32 layout (VGPR r: lanes0-15 M=r, lanes16-31 M=r+8)
#pragma unroll
  for (int t = 0; t < 2; ++t) {
    const v8f acc = t ? acc1 : acc0;
    const int nn = n0 + (wn2 + t) * 16 + ln16;
    const long nb = nn / NB, nr = nn % NB;
#pragma unroll
    for (int r = 0; r < 8; ++r) {
      const int mm = m0 + wm * 16 + r + g * 8;
      if (mm < M) {
        float val = acc[r];
        if (epilogue == 1) val = softplus_f(val + bias[mm]);
        C[(long)mm * outStrideM + nb * outBatch + nr * outStrideN] = val;
      }
    }
  }
}

// --------------------------- conv + silu (dir-aware) -----------------------
__global__ __launch_bounds__(256) void mamba_conv_silu(
    const float* __restrict__ xz, const float* __restrict__ w,
    const float* __restrict__ bias, float* __restrict__ xc, int reverse) {
  const long idx = (long)blockIdx.x * 256 + threadIdx.x;  // B*DINNER*LSEQ
  const int  l = (int)(idx % LSEQ);
  const long rest = idx / LSEQ;
  const int  d = (int)(rest % DINNER);
  const int  b = (int)(rest / DINNER);

  const float* src = xz + ((long)b * 2 * DINNER + d) * LSEQ;
  float acc = bias[d];
#pragma unroll
  for (int j = 0; j < DCONV; ++j) {
    const int li = l - (DCONV - 1) + j;
    if (li >= 0) {
      const int ls = reverse ? (LSEQ - 1 - li) : li;
      acc += w[d * DCONV + j] * src[ls];
    }
  }
  acc = acc / (1.0f + __expf(-acc));  // silu
  xc[idx] = acc;
}

// ------------------------------ selective scan -----------------------------
#define SCH 64

__global__ __launch_bounds__(256) void mamba_scan(
    const float* __restrict__ u,     // xc   [B][DINNER][L]
    const float* __restrict__ dt,    // dt   [B][DINNER][L]
    const float* __restrict__ xdbl,  // xdbl [B][96][L] (rows 64..79=B, 80..95=C)
    const float* __restrict__ xz,    // z rows DINNER..2*DINNER
    const float* __restrict__ A_log, // [DINNER][16]
    const float* __restrict__ Dv,    // [DINNER]
    float* __restrict__ y,           // [B][DINNER][L]
    int reverse) {
  __shared__ float Bs[DSTATE][SCH];
  __shared__ float Cs[DSTATE][SCH];

  const int tid = threadIdx.x;
  const int b   = blockIdx.x / (DINNER / 256);
  const int d   = (blockIdx.x % (DINNER / 256)) * 256 + tid;

  const long chBase = ((long)b * DINNER + d) * LSEQ;
  const float* up  = u + chBase;
  const float* dtp = dt + chBase;
  const float* zp  = xz + ((long)b * 2 * DINNER + DINNER + d) * LSEQ;
  float* yp = y + chBase;

  float a[DSTATE], h[DSTATE];
#pragma unroll
  for (int n = 0; n < DSTATE; ++n) {
    a[n] = -__expf(A_log[d * DSTATE + n]);
    h[n] = 0.0f;
  }
  const float Dd = Dv[d];

  const int estart = tid * 4;             // 256 threads * 4 = 16*64 elements
  const int sn = estart / SCH;
  const int sl = estart % SCH;

  for (int l0 = 0; l0 < LSEQ; l0 += SCH) {
    __syncthreads();  // previous chunk fully consumed
    {
      const float4 vb = *(const float4*)(xdbl +
          ((long)b * XDBLROWS + DTRANK + sn) * LSEQ + l0 + sl);
      *(float4*)&Bs[sn][sl] = vb;
      const float4 vc = *(const float4*)(xdbl +
          ((long)b * XDBLROWS + DTRANK + DSTATE + sn) * LSEQ + l0 + sl);
      *(float4*)&Cs[sn][sl] = vc;
    }
    __syncthreads();

    if (l0 + SCH < LSEQ) {  // prefetch next dt/u chunk (global_prefetch_b8)
      __builtin_prefetch(dtp + l0 + SCH, 0, 1);
      __builtin_prefetch(up + l0 + SCH, 0, 1);
    }

    for (int lc = 0; lc < SCH; lc += 4) {
      const float4 dt4 = *(const float4*)(dtp + l0 + lc);
      const float4 u4  = *(const float4*)(up + l0 + lc);
#pragma unroll
      for (int s = 0; s < 4; ++s) {
        const float dtl = (&dt4.x)[s];
        const float ul  = (&u4.x)[s];
        const float du  = dtl * ul;
        float ys = 0.0f;
#pragma unroll
        for (int n = 0; n < DSTATE; ++n) {
          h[n] = __expf(dtl * a[n]) * h[n] + du * Bs[n][lc + s];
          ys += h[n] * Cs[n][lc + s];
        }
        const int l  = l0 + lc + s;
        const int lz = reverse ? (LSEQ - 1 - l) : l;
        const float zv  = zp[lz];
        const float out = (ys + Dd * ul) * (zv / (1.0f + __expf(-zv)));
        if (reverse) {
          yp[LSEQ - 1 - l] += out;
        } else {
          yp[l] = out;
        }
      }
    }
  }
}

// ------------------------------- launcher ----------------------------------
static inline void launch_cvt(const float* src, __bf16* dst, long n, long npad,
                              hipStream_t stream) {
  const long n4 = npad / 4;
  hipLaunchKernelGGL(mamba_cvt_bf16,
                     dim3((unsigned)((n4 + 255) / 256)), dim3(256), 0, stream,
                     src, dst, n / 4, n4);
}

extern "C" void kernel_launch(void* const* d_in, const int* in_sizes, int n_in,
                              void* d_out, int out_size, void* d_ws, size_t ws_size,
                              hipStream_t stream) {
  const float* hs        = (const float*)d_in[0];
  const float* in_proj_w = (const float*)d_in[1];
  const float* conv_w[2]   = {(const float*)d_in[2],  (const float*)d_in[9]};
  const float* conv_b[2]   = {(const float*)d_in[3],  (const float*)d_in[10]};
  const float* x_proj_w[2] = {(const float*)d_in[4],  (const float*)d_in[11]};
  const float* dt_proj_w[2]= {(const float*)d_in[5],  (const float*)d_in[12]};
  const float* dt_proj_b[2]= {(const float*)d_in[6],  (const float*)d_in[13]};
  const float* A_log[2]    = {(const float*)d_in[7],  (const float*)d_in[14]};
  const float* Dvec[2]     = {(const float*)d_in[8],  (const float*)d_in[15]};
  const float* out_proj_w  = (const float*)d_in[16];
  float* out = (float*)d_out;

  // workspace layout
  float* ws   = (float*)d_ws;
  float* xz   = ws;                                    // [B][2*DINNER][L]
  float* xc   = xz   + (long)BDIM * 2 * DINNER * LSEQ; // [B][DINNER][L]
  float* xdbl = xc   + (long)BDIM * DINNER * LSEQ;     // [B][96][L]
  float* dtb  = xdbl + (long)BDIM * XDBLROWS * LSEQ;   // [B][DINNER][L]
  float* y    = dtb  + (long)BDIM * DINNER * LSEQ;     // [B][DINNER][L]
  __bf16* bfp = (__bf16*)(y + (long)BDIM * DINNER * LSEQ);
  __bf16* w_in_bf  = bfp;                     bfp += (long)2 * DINNER * DMODEL;
  __bf16* w_x_bf[2]  = {bfp, bfp + (long)XDBLPAD * DINNER};   // padded 128 rows
  bfp += (long)2 * XDBLPAD * DINNER;
  __bf16* w_dt_bf[2] = {bfp, bfp + (long)DINNER * DTRANK};
  bfp += (long)2 * DINNER * DTRANK;
  __bf16* w_out_bf = bfp;

  // 0) weight conversions (f32 -> bf16; x_proj zero-padded to 128 rows)
  launch_cvt(in_proj_w, w_in_bf,
             (long)2 * DINNER * DMODEL, (long)2 * DINNER * DMODEL, stream);
  launch_cvt(x_proj_w[0],  w_x_bf[0],
             (long)XDBLROWS * DINNER, (long)XDBLPAD * DINNER, stream);
  launch_cvt(x_proj_w[1],  w_x_bf[1],
             (long)XDBLROWS * DINNER, (long)XDBLPAD * DINNER, stream);
  launch_cvt(dt_proj_w[0], w_dt_bf[0],
             (long)DINNER * DTRANK, (long)DINNER * DTRANK, stream);
  launch_cvt(dt_proj_w[1], w_dt_bf[1],
             (long)DINNER * DTRANK, (long)DINNER * DTRANK, stream);
  launch_cvt(out_proj_w, w_out_bf,
             (long)DMODEL * DINNER, (long)DMODEL * DINNER, stream);

  const int NTOT = BDIM * LSEQ;  // 4096 GEMM columns

  // 1) in_proj: xz = W_in(4096x1024) . hs^T
  {
    dim3 grid(2 * DINNER / BM, NTOT / BN);
    hipLaunchKernelGGL(mamba_gemm_wmma, grid, dim3(256), 0, stream,
        w_in_bf,
        hs, (long)LSEQ * DMODEL, (long)DMODEL, 1L,
        xz, (long)2 * DINNER * LSEQ, (long)LSEQ, 1L,
        2 * DINNER, NTOT, DMODEL, LSEQ, 0, (const float*)nullptr);
  }

  for (int dir = 0; dir < 2; ++dir) {
    // 2) causal conv + silu (time-reversed input for dir==1)
    {
      const long total = (long)BDIM * DINNER * LSEQ;
      hipLaunchKernelGGL(mamba_conv_silu, dim3((unsigned)(total / 256)),
                         dim3(256), 0, stream,
                         xz, conv_w[dir], conv_b[dir], xc, dir);
    }
    // 3) x_proj: xdbl = W_x(96x2048, padded to 128) . xc
    {
      dim3 grid(XDBLPAD / BM, NTOT / BN);
      hipLaunchKernelGGL(mamba_gemm_wmma, grid, dim3(256), 0, stream,
          w_x_bf[dir],
          xc, (long)DINNER * LSEQ, 1L, (long)LSEQ,
          xdbl, (long)XDBLROWS * LSEQ, (long)LSEQ, 1L,
          XDBLROWS, NTOT, DINNER, LSEQ, 0, (const float*)nullptr);
    }
    // 4) dt: dt = softplus(W_dt(2048x64) . dtv + bias)
    {
      dim3 grid(DINNER / BM, NTOT / BN);
      hipLaunchKernelGGL(mamba_gemm_wmma, grid, dim3(256), 0, stream,
          w_dt_bf[dir],
          xdbl, (long)XDBLROWS * LSEQ, 1L, (long)LSEQ,
          dtb, (long)DINNER * LSEQ, (long)LSEQ, 1L,
          DINNER, NTOT, DTRANK, LSEQ, 1, dt_proj_b[dir]);
    }
    // 5) selective scan (+ silu(z) gate); dir==1 accumulates reversed
    {
      dim3 grid(BDIM * (DINNER / 256));
      hipLaunchKernelGGL(mamba_scan, grid, dim3(256), 0, stream,
          xc, dtb, xdbl, xz, A_log[dir], Dvec[dir], y, dir);
    }
  }

  // 6) out_proj: out[b,l,o] = W_out(1024x2048) . y
  {
    dim3 grid(DMODEL / BM, NTOT / BN);
    hipLaunchKernelGGL(mamba_gemm_wmma, grid, dim3(256), 0, stream,
        w_out_bf,
        y, (long)DINNER * LSEQ, 1L, (long)LSEQ,
        out, (long)LSEQ * DMODEL, 1L, (long)DMODEL,
        DMODEL, NTOT, DINNER, LSEQ, 0, (const float*)nullptr);
  }
}
